// SelfAttention_56590489092422
// MI455X (gfx1250) — matrix operand order repaired and checked by reading, not compiled, not run on hardware
//
#include <hip/hip_runtime.h>
#include <hip/hip_bf16.h>

#define S_LEN 4096
#define DMODEL 768
#define NHEAD 12
#define HDIM 64
#define VT_PITCH 40   // V^T tile row pitch (80B: 16B-aligned frag reads, spreads banks)
#define KT_PITCH 72   // K tile row pitch (144B = 128B row + 16B TDM pad)
#define WT_PITCH 40   // W tile row pitch (80B = 64B row + 16B TDM pad)

#if defined(__HIP_DEVICE_COMPILE__) && __has_builtin(__builtin_amdgcn_tensor_load_to_lds)
#define HAVE_TDM 1
#else
#define HAVE_TDM 0
#endif

typedef __attribute__((ext_vector_type(16))) __bf16 v16bf;
typedef __attribute__((ext_vector_type(8)))  float  v8f;
typedef __attribute__((ext_vector_type(4)))  __bf16 v4bf;
typedef __attribute__((ext_vector_type(4)))  unsigned int u32x4;
typedef __attribute__((ext_vector_type(8)))  int i32x8;
typedef __attribute__((ext_vector_type(4)))  int i32x4;

// ---------------------------------------------------------------------------
// WMMA helper: D = A(16x32 bf16) * B(32x16 bf16) + C(16x16 f32)
// ---------------------------------------------------------------------------
__device__ __forceinline__ v8f wmma_bf16(v16bf a, v16bf b, v8f c) {
  return __builtin_amdgcn_wmma_f32_16x16x32_bf16(
      /*neg_a=*/false, a, /*neg_b=*/false, b,
      /*c_mod=*/(short)0, c, /*reuse_a=*/false, /*reuse_b=*/false);
}

// 16 rows x 32 k fragment: lane&15 selects row, lane>>4 selects 16-elem k-half.
__device__ __forceinline__ v16bf load_frag16(const __bf16* base, int ld, int lane) {
  const __bf16* p = base + (size_t)(lane & 15) * ld + ((lane >> 4) << 4);
  v16bf r;
  ((float4*)&r)[0] = ((const float4*)p)[0];
  ((float4*)&r)[1] = ((const float4*)p)[1];
  return r;
}

// 32 rows x 16 cols fragment: lane selects row, 16 contiguous elems per lane.
__device__ __forceinline__ v16bf load_frag32r(const __bf16* base, int ld, int lane) {
  const __bf16* p = base + (size_t)lane * ld;
  v16bf r;
  ((float4*)&r)[0] = ((const float4*)p)[0];
  ((float4*)&r)[1] = ((const float4*)p)[1];
  return r;
}

#if HAVE_TDM
// Issue a 2D TDM load: tile_dim1 rows x tile-row of row_bytes, row stride in
// global = stride elems (bf16), LDS padded per descriptor g1 (built by caller).
__device__ __forceinline__ void tdm_load_2d(const __bf16* gsrc, unsigned ldsoff,
                                            i32x8 g1) {
  unsigned long long ga = (unsigned long long)(size_t)gsrc;
  u32x4 g0 = { 1u,                        // count=1, user descriptor
               ldsoff,                     // lds_addr (bytes)
               (unsigned)ga,               // global_addr[31:0]
               (unsigned)((ga >> 32) & 0x1FFFFFFu) | (2u << 30) };  // type=2
  const i32x4 z4 = { 0, 0, 0, 0 };
#if __clang_major__ >= 23
  const i32x8 z8 = { 0, 0, 0, 0, 0, 0, 0, 0 };
  __builtin_amdgcn_tensor_load_to_lds(g0, g1, z4, z4, z8, 0);
#else
  __builtin_amdgcn_tensor_load_to_lds(g0, g1, z4, z4, 0);
#endif
}
#endif

// ---------------------------------------------------------------------------
// fp32 -> bf16 conversion (bandwidth-bound, vectorized by 4)
// ---------------------------------------------------------------------------
__global__ void cvt_f32_bf16(const float* __restrict__ in,
                             __bf16* __restrict__ out, int n) {
  int i = (blockIdx.x * blockDim.x + threadIdx.x) * 4;
  if (i + 3 < n) {
    float4 f = *(const float4*)(in + i);
    v4bf o;
    o[0] = (__bf16)f.x; o[1] = (__bf16)f.y;
    o[2] = (__bf16)f.z; o[3] = (__bf16)f.w;
    *(v4bf*)(out + i) = o;
  }
}

// ---------------------------------------------------------------------------
// C[M,N] = scale * (A[M,K] @ W[N,K]^T + bias[N])   (torch Linear semantics)
// block = 128 threads (4 waves); block tile 128x64; wave tile 32x64.
// W tile (64 rows x 32 k, shared by all waves) is double-buffered into LDS by
// the Tensor Data Mover; DMA of tile i+1 overlaps compute of tile i.
// ---------------------------------------------------------------------------
template <typename OutT>
__global__ void gemm_xwT(const __bf16* __restrict__ A,
                         const __bf16* __restrict__ W,
                         const float* __restrict__ bias,
                         OutT* __restrict__ C,
                         int M, int N, int K, float scaleOut) {
  const int lane = threadIdx.x & 31;
  const int wave = threadIdx.x >> 5;
  const int m0 = blockIdx.x * 128 + wave * 32;
  const int n0 = blockIdx.y * 64;

  v8f acc0[4] = {};
  v8f acc1[4] = {};
  const __bf16* arow0 = A + (size_t)m0 * K;
  const __bf16* arow1 = A + (size_t)(m0 + 16) * K;

#if HAVE_TDM
  __shared__ __bf16 wt[2][64 * WT_PITCH];
  // data_size=1(2B), pad_enable, pad_interval=3 (16 DW = 64B row),
  // pad_amount=3 (4 DW = 16B); dims: 32 x 64 tile, dim0_stride = K
  const i32x8 wg1 = { (int)0x06D10000, 32 << 16, 64 << 16, 32 << 16,
                      64, K, 0, 0 };
  const unsigned wtoff0 = (unsigned)(size_t)(&wt[0][0]);
  const unsigned wtoff1 = (unsigned)(size_t)(&wt[1][0]);
  if (wave == 0) tdm_load_2d(W + (size_t)n0 * K, wtoff0, wg1);
#endif

  int it = 0;
  for (int k0 = 0; k0 < K; k0 += 32, ++it) {
    __builtin_prefetch(arow0 + (size_t)(lane & 15) * K + k0 + 64, 0, 1);
    v16bf a0 = load_frag16(arow0 + k0, K, lane);
    v16bf a1 = load_frag16(arow1 + k0, K, lane);

#if HAVE_TDM
    const int cur = it & 1;
    if (wave == 0) __builtin_amdgcn_s_wait_tensorcnt(0);
    __syncthreads();                       // publish wt[cur]
    if (wave == 0 && k0 + 32 < K)          // prefetch next W tile via DMA
      tdm_load_2d(W + (size_t)n0 * K + k0 + 32,
                  cur ? wtoff0 : wtoff1, wg1);
#pragma unroll
    for (int f = 0; f < 4; ++f) {
      v16bf b = load_frag16(&wt[cur][f * 16 * WT_PITCH], WT_PITCH, lane);
      acc0[f] = wmma_bf16(a0, b, acc0[f]);
      acc1[f] = wmma_bf16(a1, b, acc1[f]);
    }
#else
#pragma unroll
    for (int f = 0; f < 4; ++f) {
      v16bf b = load_frag16(W + (size_t)(n0 + f * 16) * K + k0, K, lane);
      acc0[f] = wmma_bf16(a0, b, acc0[f]);
      acc1[f] = wmma_bf16(a1, b, acc1[f]);
    }
#endif
  }

  const int col = lane & 15;
  const int rowoff = (lane >> 4) << 3;
#pragma unroll
  for (int f = 0; f < 4; ++f) {
    int n = n0 + f * 16 + col;
    float b = bias[n];
#pragma unroll
    for (int r = 0; r < 8; ++r) {
      int m = m0 + rowoff + r;
      C[(size_t)m * N + n] = (OutT)((acc0[f][r] + b) * scaleOut);
      C[(size_t)(m + 16) * N + n] = (OutT)((acc1[f][r] + b) * scaleOut);
    }
  }
}

// ---------------------------------------------------------------------------
// Flash attention (transposed-score formulation, double-buffered tiles).
//   S^T = K @ Q^T : query = lane, key = VGPR row -> softmax reductions are
//                   register-wise + ONE shfl_xor(16)
//   O^T = V^T @ P^T : alpha rescale is a uniform per-lane scalar
// K tile: TDM DMA into LDS (144B padded pitch). V tile: transposed staging by
// the block. Both double-buffered: tile i+1 moves while tile i computes; one
// barrier per iteration. P^T round-trips per-wave LDS for C->B layout.
// ---------------------------------------------------------------------------
__global__ void flash_attn(const __bf16* __restrict__ Qb,
                           const __bf16* __restrict__ Kb,
                           const __bf16* __restrict__ Vb,
                           __bf16* __restrict__ Ob) {
  const int lane = threadIdx.x & 31;
  const int wave = threadIdx.x >> 5;
  const int tid  = threadIdx.x;
  const int q0 = blockIdx.x * 64 + wave * 16;
  const int hc = blockIdx.y * HDIM;

  __shared__ __bf16 kt[2][32 * KT_PITCH];    // K tile [key][d], TDM-padded
  __shared__ __bf16 vt[2][HDIM * VT_PITCH];  // V^T tile [d][key]
  __shared__ __bf16 pls[4][32 * 16];         // P per wave [query][key]

  // Q fragments (B operand of K@Q^T): 16 queries x 64 dims
  v16bf qf0 = load_frag16(Qb + (size_t)q0 * DMODEL + hc, DMODEL, lane);
  v16bf qf1 = load_frag16(Qb + (size_t)q0 * DMODEL + hc + 32, DMODEL, lane);

  float m = -1e30f, l = 0.0f;
  v8f acc[4] = {};                           // O^T: 64 d-rows x 16 queries

  const int q    = lane & 15;                // this lane's query column
  const int koff = (lane >> 4) << 3;         // key row offset of this lane-half

  // cooperative V staging: thread -> (key row, 16-dim segment)
  const int vkey = tid >> 2;                 // 0..31
  const int vd   = (tid & 3) << 4;           // 0,16,32,48

  auto stage_v = [&](int buf, int t0) {
    const __bf16* vp = Vb + (size_t)(t0 + vkey) * DMODEL + hc + vd;
    v16bf vrow;
    ((float4*)&vrow)[0] = ((const float4*)vp)[0];
    ((float4*)&vrow)[1] = ((const float4*)vp)[1];
#pragma unroll
    for (int e = 0; e < 16; ++e)
      vt[buf][(vd + e) * VT_PITCH + vkey] = vrow[e];
  };

#if HAVE_TDM
  // data_size=1(2B), pad_enable, pad_interval=4 (32 DW = 128B row),
  // pad_amount=3 (4 DW = 16B); dims: 64 x 32 tile, dim0_stride = DMODEL
  const i32x8 kg1 = { (int)0x07110000, 64 << 16, 32 << 16, 64 << 16,
                      32, DMODEL, 0, 0 };
  const unsigned ktoff0 = (unsigned)(size_t)(&kt[0][0]);
  const unsigned ktoff1 = (unsigned)(size_t)(&kt[1][0]);
  if (wave == 0) tdm_load_2d(Kb + hc, ktoff0, kg1);   // t0 = 0
#endif
  stage_v(0, 0);

  int it = 0;
  for (int t0 = 0; t0 < S_LEN; t0 += 32, ++it) {
    const int cur = it & 1;
#if HAVE_TDM
    if (wave == 0) __builtin_amdgcn_s_wait_tensorcnt(0);
#endif
    __syncthreads();                         // publish kt[cur], vt[cur]

    if (t0 + 32 < S_LEN) {                   // start moving the next tile
#if HAVE_TDM
      if (wave == 0)
        tdm_load_2d(Kb + (size_t)(t0 + 32) * DMODEL + hc,
                    cur ? ktoff0 : ktoff1, kg1);
#endif
      stage_v(cur ^ 1, t0 + 32);
    }

    // ---- S^T = K @ Q^T : two 16(key)x16(query) tiles ----
    v8f s0 = {}, s1 = {};
    {
#if HAVE_TDM
      const __bf16* kbase = &kt[cur][0];
      v16bf k00 = load_frag16(kbase, KT_PITCH, lane);
      v16bf k01 = load_frag16(kbase + 32, KT_PITCH, lane);
      v16bf k10 = load_frag16(kbase + 16 * KT_PITCH, KT_PITCH, lane);
      v16bf k11 = load_frag16(kbase + 16 * KT_PITCH + 32, KT_PITCH, lane);
#else
      v16bf k00 = load_frag16(Kb + (size_t)t0 * DMODEL + hc, DMODEL, lane);
      v16bf k01 = load_frag16(Kb + (size_t)t0 * DMODEL + hc + 32, DMODEL, lane);
      v16bf k10 = load_frag16(Kb + (size_t)(t0 + 16) * DMODEL + hc, DMODEL, lane);
      v16bf k11 = load_frag16(Kb + (size_t)(t0 + 16) * DMODEL + hc + 32, DMODEL, lane);
#endif
      s0 = wmma_bf16(k00, qf0, s0);
      s0 = wmma_bf16(k01, qf1, s0);
      s1 = wmma_bf16(k10, qf0, s1);
      s1 = wmma_bf16(k11, qf1, s1);
    }

    // ---- online softmax: register-wise reduction + one xor-16 combine ----
    float tmax = fmaxf(s0[0], s1[0]);
#pragma unroll
    for (int r = 1; r < 8; ++r) tmax = fmaxf(tmax, fmaxf(s0[r], s1[r]));
    tmax = fmaxf(tmax, __shfl_xor(tmax, 16, 32));
    const float mnew = fmaxf(m, tmax);
    const float alpha = __expf(m - mnew);
    m = mnew;

    float p[16];
    float sum = 0.0f;
#pragma unroll
    for (int r = 0; r < 8; ++r) {
      p[r]     = __expf(s0[r] - mnew);
      p[8 + r] = __expf(s1[r] - mnew);
      sum += p[r] + p[8 + r];
    }
    sum += __shfl_xor(sum, 16, 32);
    l = l * alpha + sum;

    // ---- write P [query][key] to this wave's LDS scratch ----
#pragma unroll
    for (int r = 0; r < 8; ++r) {
      pls[wave][q * 32 + koff + r]      = (__bf16)p[r];
      pls[wave][q * 32 + 16 + koff + r] = (__bf16)p[8 + r];
    }
    asm volatile("s_wait_dscnt 0" ::: "memory");  // same-wave LDS RAW

    // P^T as the 32x16 B fragment (lane&15 = query col, 16 keys contiguous per lane-half)
    v16bf pf = load_frag16(&pls[wave][0], 32, lane);

    // ---- O^T = alpha*O^T + V^T @ P^T ----
#pragma unroll
    for (int f = 0; f < 4; ++f) {
      v16bf vtf = load_frag16(&vt[cur][f * 16 * VT_PITCH], VT_PITCH, lane);
#pragma unroll
      for (int r = 0; r < 8; ++r) acc[f][r] *= alpha;
      acc[f] = wmma_bf16(vtf, pf, acc[f]);
    }
  }

  // ---- epilogue: O[q][d] = O^T / l, bf16 [S, DMODEL] ----
  const float rl = 1.0f / l;
#pragma unroll
  for (int f = 0; f < 4; ++f) {
#pragma unroll
    for (int r = 0; r < 8; ++r) {
      int d = f * 16 + koff + r;
      Ob[(size_t)(q0 + q) * DMODEL + hc + d] = (__bf16)(acc[f][r] * rl);
    }
  }
}

// ---------------------------------------------------------------------------
// launch
// ---------------------------------------------------------------------------
extern "C" void kernel_launch(void* const* d_in, const int* in_sizes, int n_in,
                              void* d_out, int out_size, void* d_ws, size_t ws_size,
                              hipStream_t stream) {
  const float* x  = (const float*)d_in[0];
  const float* wq = (const float*)d_in[1];
  const float* bq = (const float*)d_in[2];
  const float* wk = (const float*)d_in[3];
  const float* bk = (const float*)d_in[4];
  const float* wv = (const float*)d_in[5];
  const float* bv = (const float*)d_in[6];
  const float* wo = (const float*)d_in[7];
  const float* bo = (const float*)d_in[8];
  float* out = (float*)d_out;

  const size_t nx = (size_t)S_LEN * DMODEL;     // 3,145,728
  const size_t nw = (size_t)DMODEL * DMODEL;    //   589,824

  char* ws = (char*)d_ws;
  __bf16* xb  = (__bf16*)ws; ws += nx * 2;
  __bf16* wqb = (__bf16*)ws; ws += nw * 2;
  __bf16* wkb = (__bf16*)ws; ws += nw * 2;
  __bf16* wvb = (__bf16*)ws; ws += nw * 2;
  __bf16* wob = (__bf16*)ws; ws += nw * 2;
  __bf16* qb  = (__bf16*)ws; ws += nx * 2;
  __bf16* kb  = (__bf16*)ws; ws += nx * 2;
  __bf16* vb  = (__bf16*)ws; ws += nx * 2;
  __bf16* ab  = (__bf16*)ws; ws += nx * 2;     // ~36 MB total

  // 1) downconvert activations + weights to bf16
  cvt_f32_bf16<<<dim3((unsigned)(nx / 1024)), 256, 0, stream>>>(x, xb, (int)nx);
  cvt_f32_bf16<<<dim3((unsigned)(nw / 1024)), 256, 0, stream>>>(wq, wqb, (int)nw);
  cvt_f32_bf16<<<dim3((unsigned)(nw / 1024)), 256, 0, stream>>>(wk, wkb, (int)nw);
  cvt_f32_bf16<<<dim3((unsigned)(nw / 1024)), 256, 0, stream>>>(wv, wvb, (int)nw);
  cvt_f32_bf16<<<dim3((unsigned)(nw / 1024)), 256, 0, stream>>>(wo, wob, (int)nw);

  // 2) Q/K/V projections (Q pre-scaled by 1/sqrt(64) = 0.125)
  dim3 gg(S_LEN / 128, DMODEL / 64);  // 32 x 12
  gemm_xwT<__bf16><<<gg, 128, 0, stream>>>(xb, wqb, bq, qb, S_LEN, DMODEL, DMODEL, 0.125f);
  gemm_xwT<__bf16><<<gg, 128, 0, stream>>>(xb, wkb, bk, kb, S_LEN, DMODEL, DMODEL, 1.0f);
  gemm_xwT<__bf16><<<gg, 128, 0, stream>>>(xb, wvb, bv, vb, S_LEN, DMODEL, DMODEL, 1.0f);

  // 3) fused flash attention (per head)
  flash_attn<<<dim3(S_LEN / 64, NHEAD), 128, 0, stream>>>(qb, kb, vb, ab);

  // 4) output projection -> fp32 d_out
  gemm_xwT<float><<<gg, 128, 0, stream>>>(ab, wob, bo, out, S_LEN, DMODEL, DMODEL, 1.0f);
}
